// TransformerBlock_37864431681703
// MI455X (gfx1250) — matrix operand order repaired and checked by reading, not compiled, not run on hardware
//
#include <hip/hip_runtime.h>

// MI455X (gfx1250, wave32) transformer block.
// All GEMMs + attention contractions via v_wmma_f32_16x16x32_bf16 (fp32 acc).
// GEMM: 256x128 block tile, 8 waves x (64x64) = 16 WMMA tiles/wave per K-step,
// register double-buffered global->LDS staging + global_prefetch.
// Requires ws_size >= 168 MB.

typedef __bf16 bf16_t;
typedef __attribute__((ext_vector_type(8)))  __bf16 v8bf;
typedef __attribute__((ext_vector_type(16))) __bf16 v16bf;
typedef __attribute__((ext_vector_type(8)))  float  v8f;

#define HIDN  1024
#define NHEAD 16
#define HDIM  64
#define SEQL  2048
#define NBAT  4
#define NROWS (NBAT*SEQL)   /* 8192 */

__device__ __forceinline__ v8bf ldv8(const bf16_t* p) {
  return *reinterpret_cast<const v8bf*>(p);
}
__device__ __forceinline__ v16bf cat16(v8bf lo, v8bf hi) {
  return __builtin_shufflevector(lo, hi, 0,1,2,3,4,5,6,7,8,9,10,11,12,13,14,15);
}
__device__ __forceinline__ v16bf ldv16(const bf16_t* p) {
  return cat16(ldv8(p), ldv8(p + 8));
}
__device__ __forceinline__ v8f wmma_bf16(v16bf a, v16bf b, v8f c) {
  return __builtin_amdgcn_wmma_f32_16x16x32_bf16(false, a, false, b, (short)0, c, false, false);
}

// ---------------------------------------------------------------------------
// Weight convert + transpose: W[K][N] fp32 -> Wt[N][K] bf16 (coalesced writes)
// ---------------------------------------------------------------------------
__global__ __launch_bounds__(256) void wtrans_kernel(const float* __restrict__ w,
                                                     bf16_t* __restrict__ wt,
                                                     int K, int N) {
  long idx = (long)blockIdx.x * 256 + threadIdx.x;
  if (idx >= (long)K * N) return;
  long kk = idx % K;
  long n  = idx / K;
  wt[idx] = (bf16_t)w[kk * (long)N + n];
}

// ---------------------------------------------------------------------------
// V transpose: v[row][HIDN] bf16 -> vt[bh][d][s] bf16 (contiguous along s)
// ---------------------------------------------------------------------------
__global__ __launch_bounds__(256) void vtrans_kernel(const bf16_t* __restrict__ v,
                                                     bf16_t* __restrict__ vt) {
  long idx = (long)blockIdx.x * 256 + threadIdx.x;     // over NROWS*HIDN
  int s = (int)(idx & (SEQL - 1));
  long rest = idx >> 11;
  int d  = (int)(rest & (HDIM - 1));
  int bh = (int)(rest >> 6);
  int b = bh >> 4, h = bh & 15;
  vt[idx] = v[((long)b * SEQL + s) * HIDN + h * HDIM + d];
}

// ---------------------------------------------------------------------------
// LayerNorm: fp32 in -> bf16 out, one 256-thread block per row of 1024
// ---------------------------------------------------------------------------
__global__ __launch_bounds__(256) void ln_kernel(const float* __restrict__ x,
                                                 const float* __restrict__ g,
                                                 const float* __restrict__ bta,
                                                 bf16_t* __restrict__ y) {
  __shared__ float red[16];
  const int row = blockIdx.x;
  const int t = threadIdx.x;
  const float* xr = x + (long)row * HIDN;
  float lx[4]; float s = 0.f, ss = 0.f;
#pragma unroll
  for (int u = 0; u < 4; ++u) { float v = xr[t + 256*u]; lx[u] = v; s += v; ss += v*v; }
#pragma unroll
  for (int m = 1; m < 32; m <<= 1) { s += __shfl_xor(s, m, 32); ss += __shfl_xor(ss, m, 32); }
  if ((t & 31) == 0) { red[(t >> 5)*2] = s; red[(t >> 5)*2 + 1] = ss; }
  __syncthreads();
  s = 0.f; ss = 0.f;
#pragma unroll
  for (int wv = 0; wv < 8; ++wv) { s += red[wv*2]; ss += red[wv*2 + 1]; }
  float mean = s * (1.0f / HIDN);
  float var  = ss * (1.0f / HIDN) - mean * mean;
  float rstd = rsqrtf(var + 1e-5f);
  bf16_t* yr = y + (long)row * HIDN;
#pragma unroll
  for (int u = 0; u < 4; ++u) {
    int c = t + 256*u;
    yr[c] = (bf16_t)((lx[u] - mean) * rstd * g[c] + bta[c]);
  }
}

// ---------------------------------------------------------------------------
// GEMM: C[M][N] = A[M][K](bf16) * Wt[N][K](bf16) + bias, epilogue modes:
//   0: bf16 out, +bias        1: bf16 out, gelu(x+bias)   2: f32 out, +bias+res
// Block tile 256x128x32, 8 waves (4x2), each wave 64x64 (4x4 WMMA tiles).
// Register double-buffering of the next K-tile + prefetch two tiles ahead.
// ---------------------------------------------------------------------------
#define BM 256
#define BN 128
#define BK 32
#define LPAD 40

__global__ __launch_bounds__(256) void gemm_bf16_kernel(
    const bf16_t* __restrict__ A, const bf16_t* __restrict__ Wt,
    const float* __restrict__ bias, const float* __restrict__ res,
    float* __restrict__ outF, bf16_t* __restrict__ outB,
    int M, int N, int K, int mode) {
  __shared__ bf16_t As[BM * LPAD];   // 20 KB
  __shared__ bf16_t Bs[BN * LPAD];   // 10 KB
  const int tid  = threadIdx.x;
  const int lane = tid & 31;
  const int wave = tid >> 5;
  const int wr = wave >> 1;          // 0..3 -> 64-row group
  const int wc = wave & 1;           // 0..1 -> 64-col group
  const long bm = (long)blockIdx.y * BM;
  const long bn = (long)blockIdx.x * BN;

  v8f acc[4][4];
#pragma unroll
  for (int mt = 0; mt < 4; ++mt)
#pragma unroll
    for (int nt = 0; nt < 4; ++nt)
#pragma unroll
      for (int j = 0; j < 8; ++j) acc[mt][nt][j] = 0.f;

  // cooperative staging: A row = tid (256 rows x 32), B row = tid>>1 (128 x 32)
  const bf16_t* Arow = A  + (bm + tid) * (long)K;
  const int brow = tid >> 1;
  const int blc  = (tid & 1) * 16;
  const bf16_t* Brow = Wt + (bn + brow) * (long)K + blc;

  v8bf ra[4], rb[2];
#pragma unroll
  for (int u = 0; u < 4; ++u) ra[u] = ldv8(Arow + 8 * u);
  rb[0] = ldv8(Brow);
  rb[1] = ldv8(Brow + 8);

  const int kb8  = (lane >> 4) * 8;
  const int kb16 = (lane >> 4) * 16;
  const int ml   = lane & 15;

  for (int k0 = 0; k0 < K; k0 += BK) {
#pragma unroll
    for (int u = 0; u < 4; ++u)
      *reinterpret_cast<v8bf*>(&As[tid * LPAD + 8 * u]) = ra[u];
    *reinterpret_cast<v8bf*>(&Bs[brow * LPAD + blc])     = rb[0];
    *reinterpret_cast<v8bf*>(&Bs[brow * LPAD + blc + 8]) = rb[1];
    __syncthreads();

    const bool more = (k0 + BK) < K;
    v8bf na[4], nb[2];
    if (more) {
#pragma unroll
      for (int u = 0; u < 4; ++u) na[u] = ldv8(Arow + k0 + BK + 8 * u);
      nb[0] = ldv8(Brow + k0 + BK);
      nb[1] = ldv8(Brow + k0 + BK + 8);
      if ((k0 + 2 * BK) < K) {
        __builtin_prefetch(Arow + k0 + 2 * BK, 0, 1);
        __builtin_prefetch(Brow + k0 + 2 * BK, 0, 1);
      }
    }

    v16bf afrag[4], bfrag[4];
#pragma unroll
    for (int mt = 0; mt < 4; ++mt) {
      int row = wr * 64 + mt * 16 + ml;
      afrag[mt] = cat16(ldv8(&As[row * LPAD + kb8]), ldv8(&As[row * LPAD + kb8 + 16]));
    }
#pragma unroll
    for (int nt = 0; nt < 4; ++nt) {
      int row = wc * 64 + nt * 16 + ml;
      bfrag[nt] = cat16(ldv8(&Bs[row * LPAD + kb8]), ldv8(&Bs[row * LPAD + kb8 + 16]));
    }
#pragma unroll
    for (int mt = 0; mt < 4; ++mt)
#pragma unroll
      for (int nt = 0; nt < 4; ++nt)
        acc[mt][nt] = wmma_bf16(afrag[mt], bfrag[nt], acc[mt][nt]);
    __syncthreads();

    if (more) {
#pragma unroll
      for (int u = 0; u < 4; ++u) ra[u] = na[u];
      rb[0] = nb[0];
      rb[1] = nb[1];
    }
  }

  const int half = lane >> 4;
  const int cn   = lane & 15;
#pragma unroll
  for (int nt = 0; nt < 4; ++nt) {
    long col = bn + wc * 64 + nt * 16 + cn;
    float bv = bias[col];
#pragma unroll
    for (int mt = 0; mt < 4; ++mt) {
#pragma unroll
      for (int j = 0; j < 8; ++j) {
        long row = bm + wr * 64 + mt * 16 + j + 8 * half;
        float v = acc[mt][nt][j] + bv;
        if (mode == 1) v = 0.5f * v * (1.0f + erff(v * 0.70710678118654752f));
        if (mode == 2) outF[row * (long)N + col] = v + res[row * (long)N + col];
        else           outB[row * (long)N + col] = (bf16_t)v;
      }
    }
  }
}

// ---------------------------------------------------------------------------
// Windowed attention, per-output-region formulation.
// Region r=0: q[0,128) from window [0,384); r=8: q[1920,2048) from [1664,2048);
// r=1..7 (i=256r): q[i-128,i+128) = blend of softmax over windows
//   A=[max(0,i-384), i+128) and B=[i-128, min(2048,i+384)), alpha = t/255.
// grid = (9 regions, 64 batch*heads), 256 threads (8 waves),
// each wave flash-softmaxes 16-query tiles against the window keys.
// ---------------------------------------------------------------------------
#define PSTR 40
__global__ __launch_bounds__(256) void attn_kernel(const bf16_t* __restrict__ q,
                                                   const bf16_t* __restrict__ kmat,
                                                   const bf16_t* __restrict__ vt,
                                                   bf16_t* __restrict__ ctx) {
  __shared__ bf16_t p_lds[8 * 16 * PSTR];
  const int r  = blockIdx.x;
  const int bh = blockIdx.y;
  const int b = bh >> 4, h = bh & 15;
  const int lane = threadIdx.x & 31;
  const int wave = threadIdx.x >> 5;

  int q0, nq, ws[2], wl[2], nwin;
  if (r == 0)      { q0 = 0;    nq = 128; ws[0] = 0;    wl[0] = 384; ws[1] = 0; wl[1] = 0; nwin = 1; }
  else if (r == 8) { q0 = 1920; nq = 128; ws[0] = 1664; wl[0] = 384; ws[1] = 0; wl[1] = 0; nwin = 1; }
  else {
    int i = 256 * r;
    q0 = i - 128; nq = 256; nwin = 2;
    ws[0] = (i >= 384) ? (i - 384) : 0;
    wl[0] = (i + 128) - ws[0];
    ws[1] = i - 128;
    int we = i + 384; if (we > SEQL) we = SEQL;
    wl[1] = we - ws[1];
  }

  const int ml   = lane & 15;
  const int half = lane >> 4;
  const int kb8  = half * 8;
  bf16_t* pw = &p_lds[wave * 16 * PSTR];

  for (int qt = wave; qt * 16 < nq; qt += 8) {
    long qrow = ((long)b * SEQL + q0 + qt * 16 + ml) * HIDN + h * HDIM;
    v16bf qa0 = cat16(ldv8(q + qrow + half * 16),      ldv8(q + qrow + half * 16 + 8));       // d 0..31 (d = 16h+s)
    v16bf qa1 = cat16(ldv8(q + qrow + 32 + half * 16), ldv8(q + qrow + 32 + half * 16 + 8));  // d 32..63 (d = 32+16h+s)

    v8f Owin[2][4];
#pragma unroll
    for (int w = 0; w < 2; ++w)
#pragma unroll
      for (int dt = 0; dt < 4; ++dt)
#pragma unroll
        for (int j = 0; j < 8; ++j) Owin[w][dt][j] = 0.f;

    for (int w = 0; w < nwin; ++w) {
      v8f O[4];
      float mrun[8], lrun[8];
#pragma unroll
      for (int dt = 0; dt < 4; ++dt)
#pragma unroll
        for (int j = 0; j < 8; ++j) O[dt][j] = 0.f;
#pragma unroll
      for (int j = 0; j < 8; ++j) { mrun[j] = -1e30f; lrun[j] = 0.f; }

      for (int kk = 0; kk < wl[w]; kk += 32) {
        const int kb16 = half * 16;
        v8f S0, S1;
#pragma unroll
        for (int j = 0; j < 8; ++j) { S0[j] = 0.f; S1[j] = 0.f; }
        {
          long kr0 = ((long)b * SEQL + ws[w] + kk + ml) * HIDN + h * HDIM;
          S0 = wmma_bf16(qa0, ldv16(kmat + kr0 + kb16), S0);
          S0 = wmma_bf16(qa1, ldv16(kmat + kr0 + 32 + kb16), S0);
          long kr1 = kr0 + (long)16 * HIDN;
          S1 = wmma_bf16(qa0, ldv16(kmat + kr1 + kb16), S1);
          S1 = wmma_bf16(qa1, ldv16(kmat + kr1 + 32 + kb16), S1);
        }
        float fac[8], psum[8];
#pragma unroll
        for (int j = 0; j < 8; ++j) {
          float s0 = S0[j] * 0.125f, s1 = S1[j] * 0.125f;   // scale = HD^-0.5
          float mx = fmaxf(s0, s1);
          mx = fmaxf(mx, __shfl_xor(mx, 1, 16));
          mx = fmaxf(mx, __shfl_xor(mx, 2, 16));
          mx = fmaxf(mx, __shfl_xor(mx, 4, 16));
          mx = fmaxf(mx, __shfl_xor(mx, 8, 16));
          float mnew = fmaxf(mrun[j], mx);
          fac[j] = __expf(mrun[j] - mnew);
          mrun[j] = mnew;
          float p0 = __expf(s0 - mnew);
          float p1 = __expf(s1 - mnew);
          S0[j] = p0; S1[j] = p1;
          float ps = p0 + p1;
          ps += __shfl_xor(ps, 1, 16);
          ps += __shfl_xor(ps, 2, 16);
          ps += __shfl_xor(ps, 4, 16);
          ps += __shfl_xor(ps, 8, 16);
          psum[j] = ps;
        }
#pragma unroll
        for (int j = 0; j < 8; ++j) {
          lrun[j] = lrun[j] * fac[j] + psum[j];
#pragma unroll
          for (int dt = 0; dt < 4; ++dt) O[dt][j] *= fac[j];
        }
        // stage P (C-frag -> LDS row-major) then read back as A-frag
#pragma unroll
        for (int j = 0; j < 8; ++j) {
          int prow = j + 8 * half;
          pw[prow * PSTR + ml]      = (bf16_t)S0[j];
          pw[prow * PSTR + ml + 16] = (bf16_t)S1[j];
        }
        __syncthreads();
        v16bf pa = cat16(ldv8(&pw[ml * PSTR + half * 16]), ldv8(&pw[ml * PSTR + half * 16 + 8]));
#pragma unroll
        for (int dt = 0; dt < 4; ++dt) {
          long vr = ((long)bh * HDIM + dt * 16 + ml) * SEQL + ws[w] + kk + kb16;
          O[dt] = wmma_bf16(pa, ldv16(vt + vr), O[dt]);
        }
        __syncthreads();
      }
#pragma unroll
      for (int j = 0; j < 8; ++j) {
        float inv = 1.0f / lrun[j];
#pragma unroll
        for (int dt = 0; dt < 4; ++dt) O[dt][j] *= inv;
      }
#pragma unroll
      for (int dt = 0; dt < 4; ++dt) Owin[w][dt] = O[dt];
    }

    // blend + store
#pragma unroll
    for (int j = 0; j < 8; ++j) {
      int rloc = qt * 16 + j + 8 * half;
      long orow = ((long)b * SEQL + q0 + rloc) * HIDN + h * HDIM;
      float a = (float)rloc * (1.0f / 255.0f);
#pragma unroll
      for (int dt = 0; dt < 4; ++dt) {
        float v0 = Owin[0][dt][j];
        float val = (nwin == 2) ? ((1.0f - a) * v0 + a * Owin[1][dt][j]) : v0;
        ctx[orow + dt * 16 + ml] = (bf16_t)val;
      }
    }
  }
}

// ---------------------------------------------------------------------------
extern "C" void kernel_launch(void* const* d_in, const int* in_sizes, int n_in,
                              void* d_out, int out_size, void* d_ws, size_t ws_size,
                              hipStream_t stream) {
  (void)in_sizes; (void)n_in; (void)out_size; (void)ws_size;
  const float* hidden = (const float*)d_in[0];
  const float* ln1_g  = (const float*)d_in[1];
  const float* ln1_b  = (const float*)d_in[2];
  const float* wq     = (const float*)d_in[3];
  const float* bq     = (const float*)d_in[4];
  const float* wk     = (const float*)d_in[5];
  const float* bk     = (const float*)d_in[6];
  const float* wv     = (const float*)d_in[7];
  const float* bv     = (const float*)d_in[8];
  const float* wproj  = (const float*)d_in[9];
  const float* bproj  = (const float*)d_in[10];
  const float* ln2_g  = (const float*)d_in[11];
  const float* ln2_b  = (const float*)d_in[12];
  const float* w1     = (const float*)d_in[13];
  const float* b1     = (const float*)d_in[14];
  const float* w2     = (const float*)d_in[15];
  const float* b2     = (const float*)d_in[16];

  char* ws = (char*)d_ws;
  const size_t MB = (size_t)1 << 20;
  bf16_t* wqT  = (bf16_t*)(ws + 0 * MB);
  bf16_t* wkT  = (bf16_t*)(ws + 2 * MB);
  bf16_t* wvT  = (bf16_t*)(ws + 4 * MB);
  bf16_t* wpT  = (bf16_t*)(ws + 6 * MB);
  bf16_t* w1T  = (bf16_t*)(ws + 8 * MB);    // 4096 x 1024
  bf16_t* w2T  = (bf16_t*)(ws + 16 * MB);   // 1024 x 4096
  bf16_t* qb   = (bf16_t*)(ws + 24 * MB);
  bf16_t* kbuf = (bf16_t*)(ws + 40 * MB);
  bf16_t* vbuf = (bf16_t*)(ws + 56 * MB);
  bf16_t* xln  = (bf16_t*)(ws + 72 * MB);
  bf16_t* vtb  = (bf16_t*)(ws + 88 * MB);
  bf16_t* ctx  = (bf16_t*)(ws + 104 * MB);
  float*  attn = (float*)(ws + 120 * MB);   // 32 MB fp32
  bf16_t* h2   = (bf16_t*)(ws + 152 * MB);
  bf16_t* gbuf = (bf16_t*)(ws + 24 * MB);   // aliases q/k/v/xln (dead by FFN1)

  { long n = (long)HIDN * HIDN; int blk = (int)((n + 255) / 256);
    wtrans_kernel<<<blk, 256, 0, stream>>>(wq, wqT, HIDN, HIDN);
    wtrans_kernel<<<blk, 256, 0, stream>>>(wk, wkT, HIDN, HIDN);
    wtrans_kernel<<<blk, 256, 0, stream>>>(wv, wvT, HIDN, HIDN);
    wtrans_kernel<<<blk, 256, 0, stream>>>(wproj, wpT, HIDN, HIDN); }
  { long n = (long)HIDN * 4 * HIDN; int blk = (int)((n + 255) / 256);
    wtrans_kernel<<<blk, 256, 0, stream>>>(w1, w1T, HIDN, 4 * HIDN);
    wtrans_kernel<<<blk, 256, 0, stream>>>(w2, w2T, 4 * HIDN, HIDN); }

  ln_kernel<<<NROWS, 256, 0, stream>>>(hidden, ln1_g, ln1_b, xln);

  dim3 g1(HIDN / BN, NROWS / BM);
  gemm_bf16_kernel<<<g1, 256, 0, stream>>>(xln, wqT, bq, nullptr, nullptr, qb,   NROWS, HIDN, HIDN, 0);
  gemm_bf16_kernel<<<g1, 256, 0, stream>>>(xln, wkT, bk, nullptr, nullptr, kbuf, NROWS, HIDN, HIDN, 0);
  gemm_bf16_kernel<<<g1, 256, 0, stream>>>(xln, wvT, bv, nullptr, nullptr, vbuf, NROWS, HIDN, HIDN, 0);

  vtrans_kernel<<<(NROWS * HIDN) / 256, 256, 0, stream>>>(vbuf, vtb);

  attn_kernel<<<dim3(9, 64), 256, 0, stream>>>(qb, kbuf, vtb, ctx);

  gemm_bf16_kernel<<<g1, 256, 0, stream>>>(ctx, wpT, bproj, hidden, attn, nullptr, NROWS, HIDN, HIDN, 2);

  ln_kernel<<<NROWS, 256, 0, stream>>>(attn, ln2_g, ln2_b, h2);

  dim3 g2((4 * HIDN) / BN, NROWS / BM);
  gemm_bf16_kernel<<<g2, 256, 0, stream>>>(h2, w1T, b1, nullptr, nullptr, gbuf, NROWS, 4 * HIDN, HIDN, 1);

  gemm_bf16_kernel<<<g1, 256, 0, stream>>>(gbuf, w2T, b2, attn, (float*)d_out, nullptr, NROWS, HIDN, 4 * HIDN, 2);
}
